// CMCLoss_47794396069892
// MI455X (gfx1250) — matrix operand order repaired and checked by reading, not compiled, not run on hardware
//
#include <hip/hip_runtime.h>

typedef unsigned short u16;
typedef __attribute__((ext_vector_type(16))) __bf16 v16bf;
typedef __attribute__((ext_vector_type(8)))  float  v8f;

#define NMOD   3
#define BROWS  4096
#define DDIM   1024
#define NCHUNK 8
#define CHCOLS (BROWS / NCHUNK)      // 512 columns per chunk
#define TEMP_INV 10.0f
#define THRES    0.5f
#define M0OFF    12.0f               // fixed LSE offset: |logit| <= ~10 (cos<=1, temp=0.1)

union FragU {
  v16bf bf;
  u16   h[16];
  uint4 u[2];
};

__device__ __forceinline__ u16 f2bf(float f) {
  unsigned u = __float_as_uint(f);
  unsigned r = ((u >> 16) & 1u) + 0x7FFFu;   // round-to-nearest-even
  return (u16)((u + r) >> 16);
}

__device__ __forceinline__ void pack8(u16* h, float4 a, float4 b) {
  h[0] = f2bf(a.x); h[1] = f2bf(a.y); h[2] = f2bf(a.z); h[3] = f2bf(a.w);
  h[4] = f2bf(b.x); h[5] = f2bf(b.y); h[6] = f2bf(b.z); h[7] = f2bf(b.w);
}

// -------- per-row L2 norms (one wave per row) --------
__global__ void row_norms_kernel(const float* __restrict__ feats,
                                 float* __restrict__ norms) {
  int wave = threadIdx.x >> 5, lane = threadIdx.x & 31;
  int row = blockIdx.x * 4 + wave;
  const float* p = feats + (size_t)row * DDIM;
  float s = 0.f;
#pragma unroll
  for (int t = 0; t < DDIM / 128; ++t) {
    float4 f = *(const float4*)(p + t * 128 + lane * 4);
    s += f.x * f.x + f.y * f.y + f.z * f.z + f.w * f.w;
  }
#pragma unroll
  for (int off = 16; off > 0; off >>= 1) s += __shfl_xor(s, off);
  if (lane == 0) norms[row] = sqrtf(s);
}

// -------- one-time f32 -> *row-normalized* bf16 copy into workspace --------
__global__ void convert_bf16_kernel(const float* __restrict__ feats,
                                    const float* __restrict__ norms,
                                    u16* __restrict__ out) {
  size_t idx = ((size_t)blockIdx.x * blockDim.x + threadIdx.x) * 8;
  size_t row = idx >> 10;                       // DDIM = 1024
  float inv = 1.0f / fmaxf(norms[row], 1e-20f);
  float4 a = *(const float4*)(feats + idx);
  float4 b = *(const float4*)(feats + idx + 4);
  uint4 o;
  o.x = (unsigned)f2bf(a.x * inv) | ((unsigned)f2bf(a.y * inv) << 16);
  o.y = (unsigned)f2bf(a.z * inv) | ((unsigned)f2bf(a.w * inv) << 16);
  o.z = (unsigned)f2bf(b.x * inv) | ((unsigned)f2bf(b.y * inv) << 16);
  o.w = (unsigned)f2bf(b.z * inv) | ((unsigned)f2bf(b.w * inv) << 16);
  *(uint4*)(out + idx) = o;
}

__global__ void init_accum_kernel(float* acc) {
  if (threadIdx.x < 8) acc[threadIdx.x] = 0.f;
}

// -------- fragment loaders (ISA layouts: A 16x32 MxK, B 32x16 KxN) --------
template <bool PRE>
__device__ __forceinline__ v16bf load_fragA(const float* pf, const u16* pb, int k0) {
  FragU f;
  if constexpr (PRE) {
    f.u[0] = *(const uint4*)(pb + k0);        // K = kbase .. kbase+7
    f.u[1] = *(const uint4*)(pb + k0 + 16);   // K = kbase+16 .. kbase+23
  } else {
    float4 x0 = *(const float4*)(pf + k0);
    float4 x1 = *(const float4*)(pf + k0 + 4);
    float4 x2 = *(const float4*)(pf + k0 + 16);
    float4 x3 = *(const float4*)(pf + k0 + 20);
    pack8(f.h, x0, x1);
    pack8(f.h + 8, x2, x3);
  }
  return f.bf;
}

template <bool PRE>
__device__ __forceinline__ v16bf load_fragB(const float* pf, const u16* pb, int k0) {
  FragU f;
  if constexpr (PRE) {
    f.u[0] = *(const uint4*)(pb + k0);        // K = kbase .. kbase+7
    f.u[1] = *(const uint4*)(pb + k0 + 16);   // K = kbase+16 .. kbase+23
  } else {
    float4 x0 = *(const float4*)(pf + k0);
    float4 x1 = *(const float4*)(pf + k0 + 4);
    float4 x2 = *(const float4*)(pf + k0 + 16);
    float4 x3 = *(const float4*)(pf + k0 + 20);
    pack8(f.h, x0, x1);
    pack8(f.h + 8, x2, x3);
  }
  return f.bf;
}

// -------- fused 3-way WMMA cosine + masked fixed-offset sum-exp --------
// grid: 3 pairs * 256 strips * NCHUNK chunks; block: 32 (one wave)
// per wave: 16 rows x 512-column chunk, N=32 per iteration (two 16-col tiles)
// PRE path consumes row-normalized bf16, so WMMA outputs are cosines directly.
template <bool PRE>
__global__ void __launch_bounds__(32, 1)
cmc_fused_kernel(const float* __restrict__ featsF,
                 const u16* __restrict__ featsB,
                 const float* __restrict__ norms,
                 float4* __restrict__ partials) {
  int wid = blockIdx.x;
  int p = wid / (256 * NCHUNK);
  int rem = wid % (256 * NCHUNK);
  int strip = rem / NCHUNK;
  int chunk = rem % NCHUNK;
  int r0 = strip * 16;
  int mi = (p < 2) ? 0 : 1;        // pairs: (0,1) (0,2) (1,2)
  int mj = (p == 0) ? 1 : 2;

  int lane = threadIdx.x;
  int lo = lane & 15, hi = lane >> 4;

  const float* aF1 = featsF + ((size_t)mi * BROWS + r0 + lo) * DDIM + hi * 8;
  const float* aF2 = featsF + ((size_t)mj * BROWS + r0 + lo) * DDIM + hi * 8;
  const u16*   aB1 = featsB + ((size_t)mi * BROWS + r0 + lo) * DDIM + hi * 8;
  const u16*   aB2 = featsB + ((size_t)mj * BROWS + r0 + lo) * DDIM + hi * 8;

  float iIr[8], iJr[8];            // used only by the non-PRE fallback
  if constexpr (!PRE) {
    const float* nI = norms + mi * BROWS;
    const float* nJ = norms + mj * BROWS;
#pragma unroll
    for (int v = 0; v < 8; ++v) {
      int r = r0 + v + hi * 8;
      iIr[v] = 1.0f / fmaxf(nI[r], 1e-20f);
      iJr[v] = 1.0f / fmaxf(nJ[r], 1e-20f);
    }
  }

  float srun[8], cnt[8], diag[8];
#pragma unroll
  for (int v = 0; v < 8; ++v) { srun[v] = 0.f; cnt[v] = 0.f; diag[v] = 0.f; }

  int cbeg = chunk * CHCOLS, cend = cbeg + CHCOLS;
  for (int n0 = cbeg; n0 < cend; n0 += 32) {
    const float* bFj = featsF + ((size_t)mj * BROWS + n0 + lo) * DDIM + hi * 8;
    const float* bFi = featsF + ((size_t)mi * BROWS + n0 + lo) * DDIM + hi * 8;
    const u16*   bBj = featsB + ((size_t)mj * BROWS + n0 + lo) * DDIM + hi * 8;
    const u16*   bBi = featsB + ((size_t)mi * BROWS + n0 + lo) * DDIM + hi * 8;
    const size_t T1 = (size_t)16 * DDIM;   // second 16-column tile

    v8f accC[2]  = {{}, {}};
    v8f accS1[2] = {{}, {}};
    v8f accS2[2] = {{}, {}};
#pragma unroll 2
    for (int k0 = 0; k0 < DDIM; k0 += 32) {
      v16bf a1  = load_fragA<PRE>(aF1, aB1, k0);
      v16bf a2  = load_fragA<PRE>(aF2, aB2, k0);
      v16bf bj0 = load_fragB<PRE>(bFj,      bBj,      k0);
      v16bf bj1 = load_fragB<PRE>(bFj + T1, bBj + T1, k0);
      v16bf bi0 = load_fragB<PRE>(bFi,      bBi,      k0);
      v16bf bi1 = load_fragB<PRE>(bFi + T1, bBi + T1, k0);
      accC[0]  = __builtin_amdgcn_wmma_f32_16x16x32_bf16(false, a1, false, bj0, (short)0, accC[0],  false, false);
      accC[1]  = __builtin_amdgcn_wmma_f32_16x16x32_bf16(false, a1, false, bj1, (short)0, accC[1],  false, false);
      accS1[0] = __builtin_amdgcn_wmma_f32_16x16x32_bf16(false, a1, false, bi0, (short)0, accS1[0], false, false);
      accS1[1] = __builtin_amdgcn_wmma_f32_16x16x32_bf16(false, a1, false, bi1, (short)0, accS1[1], false, false);
      accS2[0] = __builtin_amdgcn_wmma_f32_16x16x32_bf16(false, a2, false, bj0, (short)0, accS2[0], false, false);
      accS2[1] = __builtin_amdgcn_wmma_f32_16x16x32_bf16(false, a2, false, bj1, (short)0, accS2[1], false, false);
    }

#pragma unroll
    for (int nt = 0; nt < 2; ++nt) {
      int c = n0 + nt * 16 + lo;
      if constexpr (PRE) {
        // accumulators already hold cosines: no norms, no divides
#pragma unroll
        for (int v = 0; v < 8; ++v) {
          int r = r0 + v + hi * 8;
          float cr = accC[nt][v] * TEMP_INV;
          bool kept = (accS1[nt][v] <= THRES) | (accS2[nt][v] <= THRES) | (r == c);
          if (r == c) diag[v] = cr;
          if (kept) { srun[v] += __expf(cr - M0OFF); cnt[v] += 1.f; }
        }
      } else {
        const float* nI = norms + mi * BROWS;
        const float* nJ = norms + mj * BROWS;
        float iIc = 1.0f / fmaxf(nI[c], 1e-20f);
        float iJc = 1.0f / fmaxf(nJ[c], 1e-20f);
#pragma unroll
        for (int v = 0; v < 8; ++v) {
          int r = r0 + v + hi * 8;
          float cr = accC[nt][v] * (TEMP_INV * iIr[v]) * iJc;
          float s1 = accS1[nt][v] * iIr[v] * iIc;
          float s2 = accS2[nt][v] * iJr[v] * iJc;
          bool kept = (s1 <= THRES) | (s2 <= THRES) | (r == c);
          if (r == c) diag[v] = cr;
          if (kept) { srun[v] += __expf(cr - M0OFF); cnt[v] += 1.f; }
        }
      }
    }
  }

  // combine the 16 column-owners of each row (xor stays within each half-wave)
#pragma unroll
  for (int off = 1; off < 16; off <<= 1) {
#pragma unroll
    for (int v = 0; v < 8; ++v) {
      srun[v] += __shfl_xor(srun[v], off);
      cnt[v]  += __shfl_xor(cnt[v], off);
      diag[v] += __shfl_xor(diag[v], off);
    }
  }

  if (lo == 0) {   // lanes 0 and 16 each own 8 reduced rows
#pragma unroll
    for (int v = 0; v < 8; ++v) {
      int r = r0 + v + hi * 8;
      float4 q; q.x = srun[v]; q.y = cnt[v]; q.z = diag[v]; q.w = 0.f;
      partials[((size_t)(p * NCHUNK + chunk)) * BROWS + r] = q;
    }
  }
}

// -------- merge column-chunk partials, accumulate per-pair loss --------
__global__ void merge_kernel(const float4* __restrict__ partials,
                             float* __restrict__ accum) {
  int idx = blockIdx.x * 256 + threadIdx.x;   // 0 .. 3*4096-1
  int p = idx >> 12;
  int r = idx & (BROWS - 1);
  float s = 0.f, cnt = 0.f, diag = 0.f;
#pragma unroll
  for (int ch = 0; ch < NCHUNK; ++ch) {
    float4 q = partials[((size_t)(p * NCHUNK + ch)) * BROWS + r];
    s += q.x; cnt += q.y; diag += q.z;
  }
  bool valid = cnt > 1.5f;
  float loss = valid ? (logf(s) + M0OFF - diag) : 0.f;
  atomicAdd(&accum[p], loss);
  atomicAdd(&accum[4 + p], valid ? 1.f : 0.f);
}

__global__ void finalize_kernel(const float* __restrict__ accum,
                                float* __restrict__ out) {
  if (threadIdx.x == 0) {
    float e = 0.f;
    for (int p = 0; p < 3; ++p) {
      float c = accum[4 + p];
      e += (c > 0.f) ? (accum[p] / c) : 0.f;
    }
    out[0] = e * (1.f / 3.f);
  }
}

extern "C" void kernel_launch(void* const* d_in, const int* in_sizes, int n_in,
                              void* d_out, int out_size, void* d_ws, size_t ws_size,
                              hipStream_t stream) {
  (void)in_sizes; (void)n_in; (void)out_size;
  const float* feats = (const float*)d_in[0];   // [3, 4096, 1024] f32
  float* out = (float*)d_out;
  char* ws = (char*)d_ws;

  const size_t PART_BYTES = (size_t)NMOD * NCHUNK * BROWS * 16;   // float4 partials
  float*  norms    = (float*)ws;                                  // 49152 B
  float*  accum    = (float*)(ws + 49152);                        // 8 f32
  float4* partials = (float4*)(ws + 49152 + 256);
  u16*    featsB   = (u16*)(ws + 49152 + 256 + PART_BYTES);       // 25165824 B bf16

  size_t need_pre = 49152 + 256 + PART_BYTES + (size_t)NMOD * BROWS * DDIM * 2;
  bool pre = ws_size >= need_pre;

  row_norms_kernel<<<NMOD * BROWS / 4, 128, 0, stream>>>(feats, norms);
  init_accum_kernel<<<1, 32, 0, stream>>>(accum);
  if (pre) {
    convert_bf16_kernel<<<(NMOD * BROWS * DDIM) / (256 * 8), 256, 0, stream>>>(feats, norms, featsB);
    cmc_fused_kernel<true><<<3 * 256 * NCHUNK, 32, 0, stream>>>(feats, featsB, norms, partials);
  } else {
    cmc_fused_kernel<false><<<3 * 256 * NCHUNK, 32, 0, stream>>>(feats, featsB, norms, partials);
  }
  merge_kernel<<<NMOD * BROWS / 256, 256, 0, stream>>>(partials, accum);
  finalize_kernel<<<1, 32, 0, stream>>>(accum, out);
}
